// GTEALSTMT2VLayer_1331439862434
// MI455X (gfx1250) — matrix operand order repaired and checked by reading, not compiled, not run on hardware
//
#include <hip/hip_runtime.h>
#include <hip/hip_bf16.h>

typedef __attribute__((ext_vector_type(16))) _Float16 v16h;
typedef __attribute__((ext_vector_type(8)))  _Float16 h8;
typedef __attribute__((ext_vector_type(8)))  float    v8f;

#define N_SRC 16384
#define N_DSTN 8192
#define KNB 32
#define EE  (N_DSTN * KNB)   // 262144 edges
#define TT  16
#define HH  64
#define G4  256              // 4*H
#define DIN 16               // E_IN + TH

__device__ __forceinline__ float fastrcp(float x) {
    return __builtin_amdgcn_rcpf(x);          // v_rcp_f32 (1 ulp)
}
__device__ __forceinline__ float sigf(float x) {
    return fastrcp(1.0f + __expf(-x));        // v_exp + v_rcp, no div sequence
}
__device__ __forceinline__ float tanhfast(float x) {
    // tanh(x) = 1 - 2/(exp(2x)+1); saturates correctly at +-inf
    return 1.0f - 2.0f * fastrcp(__expf(2.0f * x) + 1.0f);
}

// ---------------------------------------------------------------------------
// Kernel A: Time2Vec + two LSTMs over T=16 for a tile of 16 edges.
// 256 threads = 8 waves; waves 0-3 -> lstm_e, waves 4-7 -> lstm_a.
// Wave (wv&3) owns hid slice [(wv&3)*16, +16); its 4 WMMA N-tiles are the
// i/f/g/o gate columns for that slice, so after the 12 WMMAs each lane holds
// all four gates for its (edge, hid) states in registers -> the cell update
// runs on accumulators directly (no gate LDS round trip).
// ---------------------------------------------------------------------------
__global__ __launch_bounds__(256) void lstm_t2v_kernel(
    const float* __restrict__ edge_feats, const float* __restrict__ e_times,
    const int*   __restrict__ e_len,
    const float* __restrict__ t2v_w,  const float* __restrict__ t2v_b,
    const float* __restrict__ t2v_w0, const float* __restrict__ t2v_b0,
    const float* __restrict__ Wih_e, const float* __restrict__ Whh_e,
    const float* __restrict__ bih_e, const float* __restrict__ bhh_e,
    const float* __restrict__ Wih_a, const float* __restrict__ Whh_a,
    const float* __restrict__ bih_a, const float* __restrict__ bhh_a,
    const float* __restrict__ attn_w,
    _Float16* __restrict__ eout_h, float* __restrict__ a_ws)
{
    __shared__ alignas(16) _Float16 xtile[TT * 16 * DIN];   // [step][edge][dim]
    __shared__ alignas(16) _Float16 hbuf[2 * 16 * HH];      // [lstm][edge][hid]
    __shared__ int   lens[16];
    __shared__ float apart[16];                             // attn partials / edge

    const int tid = threadIdx.x;
    const int e0  = blockIdx.x * 16;

    // ---- Time2Vec + x tile fill: one (edge,step) pair per thread, 16 dims.
    {
        const int edge = tid >> 4, step = tid & 15;
        const float tau = e_times[(size_t)(e0 + edge) * TT + step];
        _Float16* xp = &xtile[(step * 16 + edge) * DIN];
        const float* ef = &edge_feats[((size_t)(e0 + edge) * TT + step) * 8];
        #pragma unroll
        for (int j = 0; j < 8; ++j) xp[j] = (_Float16)ef[j];
        #pragma unroll
        for (int j = 0; j < 7; ++j)
            xp[8 + j] = (_Float16)__sinf(tau * t2v_w[j] + t2v_b[j]);
        xp[15] = (_Float16)(tau * t2v_w0[0] + t2v_b0[0]);
    }
    if (tid < 16) { lens[tid] = e_len[e0 + tid]; apart[tid] = 0.0f; }

    // ---- zero h state (h0 = 0): 2048 halves, 16 per thread for tid < 128
    if (tid < 128) {
        h8 z;
        #pragma unroll
        for (int j = 0; j < 8; ++j) z[j] = (_Float16)0.0f;
        *(h8*)&hbuf[tid * 16]     = z;
        *(h8*)&hbuf[tid * 16 + 8] = z;
    }

    // ---- Per-wave mapping
    const int wv   = tid >> 5;           // wave 0..7
    const int lane = tid & 31;
    const int wl   = wv >> 2;            // which LSTM this wave serves
    const int hslice = (wv & 3) * 16;    // hid slice owned by this wave
    const int hi = lane >> 4;
    const int nn = lane & 15;
    const int hid = hslice + nn;         // this lane's hidden index
    const float* Wih = wl ? Wih_a : Wih_e;
    const float* Whh = wl ? Whh_a : Whh_e;
    const float* bih = wl ? bih_a : bih_e;
    const float* bhh = wl ? bhh_a : bhh_e;

    // ---- Persistent B fragments (weights f32 -> f16); N-tile nt = gate nt.
    v16h Bx[4], Bh0[4], Bh1[4];
    float bias[4];
    #pragma unroll
    for (int nt = 0; nt < 4; ++nt) {
        const int col = nt * 64 + hid;                // gate column (N)
        bias[nt] = bih[col] + bhh[col];
        #pragma unroll
        for (int e = 0; e < 16; ++e) {
            const int k = (e < 8) ? 8 * hi + e : 8 + 8 * hi + e;  // B element e -> K = 8*hi+e (e<8), 16+8*hi+(e-8) (e>=8)
            Bx[nt][e]  = (k < DIN) ? (_Float16)Wih[col * DIN + k] : (_Float16)0.0f;
            Bh0[nt][e] = (_Float16)Whh[col * HH + k];         // K 0..31
            Bh1[nt][e] = (_Float16)Whh[col * HH + 32 + k];    // K 32..63
        }
    }

    __syncthreads();

    // per-lane persistent state: 8 edges (r + 8*hi) x 1 hid
    int lenr[8];
    float creg[8], hlast[8];
    #pragma unroll
    for (int r = 0; r < 8; ++r) {
        lenr[r] = lens[r + 8 * hi];
        creg[r] = 0.0f; hlast[r] = 0.0f;
    }

    const int row = nn;                  // A-matrix row for this lane
    for (int step = 0; step < TT; ++step) {
        // -- A fragments (ISA layout: e<8 -> K=e+8*hi, e>=8 -> K=16+(e-8)+8*hi)
        v16h Ax, Ah0, Ah1;
        {
            const _Float16* xr = &xtile[(step * 16 + row) * DIN];
            const h8 xlo = *(const h8*)&xr[8 * hi];   // K 0..15 slice
            const _Float16* hr = &hbuf[(wl * 16 + row) * HH];
            const h8 a0 = *(const h8*)&hr[ 0 + 8 * hi];
            const h8 a1 = *(const h8*)&hr[16 + 8 * hi];
            const h8 a2 = *(const h8*)&hr[32 + 8 * hi];
            const h8 a3 = *(const h8*)&hr[48 + 8 * hi];
            #pragma unroll
            for (int e = 0; e < 8; ++e) {
                Ax[e]      = xlo[e];
                Ax[8 + e]  = (_Float16)0.0f;          // K>=16 is zero padding
                Ah0[e]     = a0[e];  Ah0[8 + e] = a1[e];
                Ah1[e]     = a2[e];  Ah1[8 + e] = a3[e];
            }
        }

        // -- gates: acc[nt] = gate nt for (edges, this lane's hid)
        v8f acc[4];
        #pragma unroll
        for (int nt = 0; nt < 4; ++nt) {
            #pragma unroll
            for (int r = 0; r < 8; ++r) acc[nt][r] = bias[nt];
            acc[nt] = __builtin_amdgcn_wmma_f32_16x16x32_f16(false, Ax,  false, Bx[nt],
                                                             (short)0, acc[nt], false, false);
            acc[nt] = __builtin_amdgcn_wmma_f32_16x16x32_f16(false, Ah0, false, Bh0[nt],
                                                             (short)0, acc[nt], false, false);
            acc[nt] = __builtin_amdgcn_wmma_f32_16x16x32_f16(false, Ah1, false, Bh1[nt],
                                                             (short)0, acc[nt], false, false);
        }
        __syncthreads();   // all A-reads of hbuf done before overwriting h

        // -- LSTM cell update straight on accumulators (gate order i,f,g,o)
        #pragma unroll
        for (int r = 0; r < 8; ++r) {
            const float gi = acc[0][r], gf = acc[1][r];
            const float gg = acc[2][r], go = acc[3][r];
            const float c = sigf(gf) * creg[r] + sigf(gi) * tanhfast(gg);
            creg[r] = c;
            const float h = sigf(go) * tanhfast(c);
            if (step < lenr[r]) hlast[r] = h;
            hbuf[(wl * 16 + (r + 8 * hi)) * HH + hid] = (_Float16)h;
        }
        __syncthreads();   // h visible for next step's A fragments
    }

    // ---- Emit e_out (f16) and a = leaky_relu(a_out . attn_w)
    if (wl == 0) {
        #pragma unroll
        for (int r = 0; r < 8; ++r)
            eout_h[(size_t)(e0 + r + 8 * hi) * HH + hid] = (_Float16)hlast[r];
    } else {
        const float aw = attn_w[hid];
        #pragma unroll
        for (int r = 0; r < 8; ++r)
            atomicAdd(&apart[r + 8 * hi], hlast[r] * aw);   // ds_add_f32
    }
    __syncthreads();
    if (tid < 16) {
        const float s = apart[tid];
        a_ws[e0 + tid] = (s > 0.0f) ? s : 0.01f * s;
    }
}

// ---------------------------------------------------------------------------
// Kernel B: per dst node (K=32 edges): sparsemax, m = relu([h_src|e_out]@W^T+b)
// via WMMA (2 M-tiles x 4 N-tiles x 4 K-chunks), alpha-weighted reduce,
// then the 1x128x64 node matmul.
// ---------------------------------------------------------------------------
__global__ __launch_bounds__(256) void gather_node_kernel(
    const float* __restrict__ node_features,
    const int*   __restrict__ edge_src,
    const float* __restrict__ edge_out_W, const float* __restrict__ edge_out_b,
    const float* __restrict__ node_W,     const float* __restrict__ node_b,
    const _Float16* __restrict__ eout_h,  const float* __restrict__ a_ws,
    float* __restrict__ out)
{
    __shared__ alignas(16) _Float16 xm[32 * 128];     // [edge][h_src(64)|e_out(64)]
    __shared__ float zbuf[32], zsh[32];
    __shared__ float cnt_s[32], S_s[32];
    __shared__ float alpha[32];
    __shared__ float tau_s;
    __shared__ float hneigh[64];

    const int tid  = threadIdx.x;
    const int dst  = blockIdx.x;
    const int ebase = dst * KNB;

    if (tid < 32) zbuf[tid] = a_ws[ebase + tid];
    if (tid < 64) hneigh[tid] = 0.0f;

    // ---- gather [h_src | e_out] tile, f16. thread -> (edge, 16-dim segment)
    {
        const int edge = tid >> 3, seg = tid & 7;
        _Float16* xp = &xm[edge * 128 + seg * 16];
        if (seg < 4) {
            const int src = edge_src[ebase + edge];
            const float* nf = &node_features[(size_t)src * 64 + seg * 16];
            #pragma unroll
            for (int j = 0; j < 16; ++j) xp[j] = (_Float16)nf[j];
        } else {
            const _Float16* ep = &eout_h[(size_t)(ebase + edge) * 64 + (seg - 4) * 16];
            *(h8*)xp       = *(const h8*)ep;
            *(h8*)(xp + 8) = *(const h8*)(ep + 8);
        }
    }
    __syncthreads();

    // ---- sparsemax over 32 logits (sort-free O(K^2) support test)
    if (tid < 32) {
        float zmax = zbuf[0];
        #pragma unroll
        for (int i = 1; i < 32; ++i) zmax = fmaxf(zmax, zbuf[i]);
        const float z = zbuf[tid] - zmax;
        float cnt = 0.0f, S = 0.0f;
        #pragma unroll
        for (int i = 0; i < 32; ++i) {
            const float zi = zbuf[i] - zmax;
            if (zi >= z) { cnt += 1.0f; S += zi; }
        }
        const bool supp = (1.0f + cnt * z) > S;       // sorted-criterion equivalent
        cnt_s[tid] = supp ? cnt : 0.0f;
        S_s[tid]   = S;
        zsh[tid]   = z;
    }
    __syncthreads();
    if (tid == 0) {
        float kk = 0.0f, Ssel = 1.0f;
        #pragma unroll
        for (int i = 0; i < 32; ++i)
            if (cnt_s[i] > kk) { kk = cnt_s[i]; Ssel = S_s[i]; }
        tau_s = (Ssel - 1.0f) * fastrcp(kk);          // kk >= 1 always
    }
    __syncthreads();
    if (tid < 32) alpha[tid] = fmaxf(zsh[tid] - tau_s, 0.0f);
    __syncthreads();

    // ---- m = relu(xm @ edge_out_W^T + b), fused with alpha-weighted reduce
    const int wv = tid >> 5, lane = tid & 31, hi = lane >> 4, nn = lane & 15;
    const int mtile = wv >> 2;                        // 0 or 1 (16 edges each)
    const int colBase = (wv & 3) * 16;                // 16 output cols per wave

    v16h Bf[4];
    #pragma unroll
    for (int c = 0; c < 4; ++c)
        #pragma unroll
        for (int e = 0; e < 16; ++e)
            Bf[c][e] = (_Float16)edge_out_W[(colBase + nn) * 128 + 32 * c + ((e < 8) ? 8 * hi + e : 8 + 8 * hi + e)];

    v8f acc;
    #pragma unroll
    for (int r = 0; r < 8; ++r) acc[r] = edge_out_b[colBase + nn];

    const _Float16* xr = &xm[(mtile * 16 + nn) * 128];  // A row = lane&15
    #pragma unroll
    for (int c = 0; c < 4; ++c) {
        const h8 lo = *(const h8*)&xr[32 * c + 8 * hi];
        const h8 hi8 = *(const h8*)&xr[32 * c + 16 + 8 * hi];
        v16h A;
        #pragma unroll
        for (int e = 0; e < 8; ++e) { A[e] = lo[e]; A[8 + e] = hi8[e]; }
        acc = __builtin_amdgcn_wmma_f32_16x16x32_f16(false, A, false, Bf[c],
                                                     (short)0, acc, false, false);
    }
    {
        float part = 0.0f;
        #pragma unroll
        for (int r = 0; r < 8; ++r) {
            const int edge = mtile * 16 + r + 8 * hi;
            part += alpha[edge] * fmaxf(acc[r], 0.0f);
        }
        atomicAdd(&hneigh[colBase + nn], part);       // ds_add_f32
    }
    __syncthreads();

    // ---- h = relu([h_dst | h_neigh] @ node_W^T + b), one row, 64 cols
    if (tid < 64) {
        const float* wr = &node_W[tid * 128];
        const float* hd = &node_features[(size_t)dst * 64];
        float s = node_b[tid];
        #pragma unroll 8
        for (int k = 0; k < 64; ++k) s += hd[k] * wr[k];
        #pragma unroll 8
        for (int k = 0; k < 64; ++k) s += hneigh[k] * wr[64 + k];
        out[(size_t)dst * 64 + tid] = fmaxf(s, 0.0f);
    }
}

// ---------------------------------------------------------------------------
extern "C" void kernel_launch(void* const* d_in, const int* in_sizes, int n_in,
                              void* d_out, int out_size, void* d_ws, size_t ws_size,
                              hipStream_t stream) {
    const float* node_features = (const float*)d_in[0];
    const float* edge_feats    = (const float*)d_in[1];
    const float* e_times       = (const float*)d_in[2];
    const int*   e_len         = (const int*)  d_in[3];
    const int*   edge_src      = (const int*)  d_in[4];
    const float* t2v_w         = (const float*)d_in[5];
    const float* t2v_b         = (const float*)d_in[6];
    const float* t2v_w0        = (const float*)d_in[7];
    const float* t2v_b0        = (const float*)d_in[8];
    const float* Wih_e         = (const float*)d_in[9];
    const float* Whh_e         = (const float*)d_in[10];
    const float* bih_e         = (const float*)d_in[11];
    const float* bhh_e         = (const float*)d_in[12];
    const float* Wih_a         = (const float*)d_in[13];
    const float* Whh_a         = (const float*)d_in[14];
    const float* bih_a         = (const float*)d_in[15];
    const float* bhh_a         = (const float*)d_in[16];
    const float* attn_w        = (const float*)d_in[17];
    const float* edge_out_W    = (const float*)d_in[18];
    const float* edge_out_b    = (const float*)d_in[19];
    const float* node_W        = (const float*)d_in[20];
    const float* node_b        = (const float*)d_in[21];

    _Float16* eout_h = (_Float16*)d_ws;                       // E*64 f16 = 32 MiB
    float*    a_ws   = (float*)((char*)d_ws + (size_t)EE * HH * sizeof(_Float16));

    lstm_t2v_kernel<<<EE / 16, 256, 0, stream>>>(
        edge_feats, e_times, e_len, t2v_w, t2v_b, t2v_w0, t2v_b0,
        Wih_e, Whh_e, bih_e, bhh_e, Wih_a, Whh_a, bih_a, bhh_a,
        attn_w, eout_h, a_ws);

    gather_node_kernel<<<N_DSTN, 256, 0, stream>>>(
        node_features, edge_src, edge_out_W, edge_out_b, node_W, node_b,
        eout_h, a_ws, (float*)d_out);
}
